// AttentionLayer_65146063946112
// MI455X (gfx1250) — matrix operand order repaired and checked by reading, not compiled, not run on hardware
//
#include <hip/hip_runtime.h>

// Problem constants (reference: B=2, S=2048, D=1024, HID=4096, H=16, DH=64)
#define BB  2
#define SS  2048
#define DDIM 1024
#define HIDD 4096
#define NH  16
#define DH  64

typedef __attribute__((ext_vector_type(8)))  _Float16 v8h;
typedef __attribute__((ext_vector_type(16))) _Float16 v16h;
typedef __attribute__((ext_vector_type(8)))  float    v8f;

__device__ __forceinline__ v8f wmma_f16(v16h a, v16h b, v8f c) {
  // D = A(16x32 f16) x B(32x16 f16) + C(16x16 f32)
  return __builtin_amdgcn_wmma_f32_16x16x32_f16(false, a, false, b, (short)0, c,
                                                false, false);
}

// A-fragment (16x32, M=lane&15): lane<16 holds K=0..15 contiguous, lane>=16 K=16..31
__device__ __forceinline__ v16h load_a_frag(const _Float16* __restrict__ p, int lane) {
  return *(const v16h*)(p + ((lane & 16) ? 16 : 0));
}

// B-fragment (32x16, N=lane&15): lane<16 holds K=0..15 contiguous, lane>=16 K=16..31
__device__ __forceinline__ v16h load_b_frag(const _Float16* __restrict__ p, int lane) {
  return *(const v16h*)(p + ((lane & 16) ? 16 : 0));
}

// ---------------- fp32 -> f16 convert ----------------
__global__ void __launch_bounds__(256)
cvt_f32_f16_kernel(const float* __restrict__ X, _Float16* __restrict__ Y, int n) {
  int i = blockIdx.x * blockDim.x + threadIdx.x;
  if (i < n) Y[i] = (_Float16)X[i];
}

// ---------------- LDS-tiled transpose + convert: W[K,N] f32 -> Wt[N,K] f16 -----------
__global__ void __launch_bounds__(256)
transpose_cvt_kernel(const float* __restrict__ W, _Float16* __restrict__ Wt,
                     int K, int N) {
  __shared__ _Float16 tile[32][33];
  const int tx = threadIdx.x, ty = threadIdx.y;         // (32, 8)
  const int n = blockIdx.x * 32 + tx;
  const int kbase = blockIdx.y * 32;
#pragma unroll
  for (int r = ty; r < 32; r += 8)
    tile[r][tx] = (_Float16)W[(size_t)(kbase + r) * N + n];
  __syncthreads();
  const int k2 = kbase + tx;
  const int nbase = blockIdx.x * 32;
#pragma unroll
  for (int r = ty; r < 32; r += 8)
    Wt[(size_t)(nbase + r) * K + k2] = tile[tx][r];
}

// ---------------- GEMM1: C = silu(A @ W + bias), f16 out ----------------
// A: [M,K] f16 row-major; Bt: [N,K] f16 (W transposed); wave tile 32M x 64N.
__global__ void __launch_bounds__(256)
gemm_silu_kernel(const _Float16* __restrict__ A, const _Float16* __restrict__ Bt,
                 const float* __restrict__ bias, _Float16* __restrict__ C,
                 int M, int N, int K) {
  const int lane = threadIdx.x & 31;
  const int wave = (blockIdx.x * blockDim.x + threadIdx.x) >> 5;
  const int ntiles = N >> 6;
  const int mt = wave / ntiles, nt = wave % ntiles;
  const int m0 = mt << 5, n0 = nt << 6;
  const int lr = lane & 15;

  const _Float16* arow0 = A + (size_t)(m0 + lr) * K;
  const _Float16* arow1 = arow0 + (size_t)16 * K;
  const _Float16* brow[4];
#pragma unroll
  for (int j = 0; j < 4; ++j) brow[j] = Bt + (size_t)(n0 + j * 16 + lr) * K;

  v8f acc0[4] = {}, acc1[4] = {};
  for (int k0 = 0; k0 < K; k0 += 32) {
    __builtin_prefetch(arow0 + k0 + 256, 0, 1);
    v16h af0 = load_a_frag(arow0 + k0, lane);
    v16h af1 = load_a_frag(arow1 + k0, lane);
#pragma unroll
    for (int j = 0; j < 4; ++j) {
      v16h bf = load_b_frag(brow[j] + k0, lane);
      acc0[j] = wmma_f16(af0, bf, acc0[j]);
      acc1[j] = wmma_f16(af1, bf, acc1[j]);
    }
  }

  const int rofs = (lane & 16) ? 8 : 0;
#pragma unroll
  for (int j = 0; j < 4; ++j) {
    const int n = n0 + j * 16 + lr;
    const float bj = bias[n];
#pragma unroll
    for (int r = 0; r < 8; ++r) {
      const float v0 = acc0[j][r] + bj;
      const float v1 = acc1[j][r] + bj;
      C[(size_t)(m0 + r + rofs) * N + n]      = (_Float16)(v0 / (1.0f + __expf(-v0)));
      C[(size_t)(m0 + 16 + r + rofs) * N + n] = (_Float16)(v1 / (1.0f + __expf(-v1)));
    }
  }
}

// ---------------- GEMM2: C = A @ W + bias, f16 out (optionally scattered as V^T) ----
__global__ void __launch_bounds__(256)
gemm_out_kernel(const _Float16* __restrict__ A, const _Float16* __restrict__ Bt,
                const float* __restrict__ bias, _Float16* __restrict__ C,
                int M, int N, int K, int vt_mode) {
  const int lane = threadIdx.x & 31;
  const int wave = (blockIdx.x * blockDim.x + threadIdx.x) >> 5;
  const int ntiles = N >> 6;
  const int mt = wave / ntiles, nt = wave % ntiles;
  const int m0 = mt << 5, n0 = nt << 6;
  const int lr = lane & 15;

  const _Float16* arow0 = A + (size_t)(m0 + lr) * K;
  const _Float16* arow1 = arow0 + (size_t)16 * K;
  const _Float16* brow[4];
#pragma unroll
  for (int j = 0; j < 4; ++j) brow[j] = Bt + (size_t)(n0 + j * 16 + lr) * K;

  v8f acc0[4] = {}, acc1[4] = {};
  for (int k0 = 0; k0 < K; k0 += 32) {
    __builtin_prefetch(arow0 + k0 + 256, 0, 1);
    v16h af0 = load_a_frag(arow0 + k0, lane);
    v16h af1 = load_a_frag(arow1 + k0, lane);
#pragma unroll
    for (int j = 0; j < 4; ++j) {
      v16h bf = load_b_frag(brow[j] + k0, lane);
      acc0[j] = wmma_f16(af0, bf, acc0[j]);
      acc1[j] = wmma_f16(af1, bf, acc1[j]);
    }
  }

  const int rofs = (lane & 16) ? 8 : 0;
#pragma unroll
  for (int j = 0; j < 4; ++j) {
    const int n = n0 + j * 16 + lr;
    const float bj = bias[n];
#pragma unroll
    for (int r = 0; r < 8; ++r) {
#pragma unroll
      for (int half = 0; half < 2; ++half) {
        const int m = m0 + half * 16 + r + rofs;
        const float v = (half ? acc1[j][r] : acc0[j][r]) + bj;
        if (vt_mode) {
          // scatter into V^T layout [B][D][S]: vt[(b*D + n)*S + s]
          C[((size_t)(m >> 11) * DDIM + n) * SS + (m & (SS - 1))] = (_Float16)v;
        } else {
          C[(size_t)m * N + n] = (_Float16)v;
        }
      }
    }
  }
}

// ---------------- Flash attention (transposed tiles), one wave = 16 query rows ------
__global__ void __launch_bounds__(256)
attn_kernel(const _Float16* __restrict__ Qh, const _Float16* __restrict__ Kh,
            const _Float16* __restrict__ Vt, float* __restrict__ Out) {
  const int lane = threadIdx.x & 31;
  const int wave = (blockIdx.x * blockDim.x + threadIdx.x) >> 5;
  const int qtiles = SS / 16;
  const int qt = wave % qtiles;
  const int bh = wave / qtiles;
  const int h = bh % NH, b = bh / NH;
  const int q0 = qt * 16;
  const int qc = lane & 15;
  const bool up = (lane & 16) != 0;
  const int bo = up ? 16 : 0;

  // Q^T B-fragments (K-dim = dh), reused across all key chunks
  const _Float16* qrow = Qh + ((size_t)(b * SS + q0 + qc)) * DDIM + h * DH;
  const v16h bq0 = *(const v16h*)(qrow + bo);
  const v16h bq1 = *(const v16h*)(qrow + 32 + bo);

  const _Float16* Kb = Kh + ((size_t)b * SS) * DDIM + h * DH;
  const _Float16* Vb = Vt + ((size_t)(b * DDIM + h * DH)) * SS;

  v8f ot[4] = {};                    // O^T tiles: dh tile d, lane = q col
  float m_run = -1e30f, l_run = 0.0f;
  const int q_glob = q0 + qc;
  const int qmax = q0 + 15;
  const int kofs = up ? 8 : 0;

  for (int kb = 0; kb <= qmax; kb += 32) {
    // S^T = K_chunk(32x64) @ Q^T(64x16): two 16x16 C tiles
    const _Float16* kr0 = Kb + (size_t)(kb + qc) * DDIM;
    const _Float16* kr1 = Kb + (size_t)(kb + 16 + qc) * DDIM;
    v8f st0 = {}, st1 = {};
    st0 = wmma_f16(load_a_frag(kr0, lane), bq0, st0);
    st0 = wmma_f16(load_a_frag(kr0 + 32, lane), bq1, st0);
    st1 = wmma_f16(load_a_frag(kr1, lane), bq0, st1);
    st1 = wmma_f16(load_a_frag(kr1 + 32, lane), bq1, st1);

    // scale + causal mask + online softmax (q is fixed per lane)
    float p0[8], p1[8];
    float mx = m_run;
#pragma unroll
    for (int r = 0; r < 8; ++r) {
      const int k0i = kb + r + kofs;
      const int k1i = k0i + 16;
      p0[r] = (k0i <= q_glob) ? st0[r] * 0.125f : -1e30f;
      p1[r] = (k1i <= q_glob) ? st1[r] * 0.125f : -1e30f;
      mx = fmaxf(mx, fmaxf(p0[r], p1[r]));
    }
    mx = fmaxf(mx, __shfl_xor(mx, 16, 32));
    const float alpha = __expf(m_run - mx);
    m_run = mx;
    float rs = 0.0f;
#pragma unroll
    for (int r = 0; r < 8; ++r) {
      p0[r] = __expf(p0[r] - mx);
      p1[r] = __expf(p1[r] - mx);
      rs += p0[r] + p1[r];
    }
    rs += __shfl_xor(rs, 16, 32);
    l_run = l_run * alpha + rs;
#pragma unroll
    for (int d = 0; d < 4; ++d)
#pragma unroll
      for (int r = 0; r < 8; ++r) ot[d][r] *= alpha;

    // Assemble P^T B-fragment (32x16, K = k-token, N = q):
    //  lower lanes need rows 0..7 (own st0) + 8..15 (upper's st0)
    //  upper lanes need rows 16..23 (lower's st1) + 24..31 (own st1)
    union U { unsigned int u[4]; v8h v; } P0, P1, RC;
#pragma unroll
    for (int r = 0; r < 4; ++r) {
      union { _Float16 f[2]; unsigned int u; } c0, c1;
      c0.f[0] = (_Float16)p0[2 * r]; c0.f[1] = (_Float16)p0[2 * r + 1];
      c1.f[0] = (_Float16)p1[2 * r]; c1.f[1] = (_Float16)p1[2 * r + 1];
      P0.u[r] = c0.u; P1.u[r] = c1.u;
    }
#pragma unroll
    for (int r = 0; r < 4; ++r)
      RC.u[r] = (unsigned int)__shfl_xor((int)(up ? P0.u[r] : P1.u[r]), 16, 32);
    const v8h lo8 = up ? RC.v : P0.v;
    const v8h hi8 = up ? P1.v : RC.v;
    const v16h pf = __builtin_shufflevector(lo8, hi8,
        0,1,2,3,4,5,6,7,8,9,10,11,12,13,14,15);

    // O^T += V^T(64 x 32) @ P^T(32 x 16): A-fragments contiguous from V^T rows
#pragma unroll
    for (int d = 0; d < 4; ++d) {
      const _Float16* vr = Vb + (size_t)(d * 16 + qc) * SS + kb;
      ot[d] = wmma_f16(load_a_frag(vr, lane), pf, ot[d]);
    }
  }

  const float inv = 1.0f / l_run;
  float* orow = Out + ((size_t)(b * SS + q0 + qc)) * DDIM + h * DH;
#pragma unroll
  for (int d = 0; d < 4; ++d)
#pragma unroll
    for (int r = 0; r < 8; ++r)
      orow[d * 16 + r + kofs] = ot[d][r] * inv;
}

// ---------------- host-side launch ----------------
extern "C" void kernel_launch(void* const* d_in, const int* in_sizes, int n_in,
                              void* d_out, int out_size, void* d_ws, size_t ws_size,
                              hipStream_t stream) {
  const float* x = (const float*)d_in[0];
  const float* w1[3] = { (const float*)d_in[1], (const float*)d_in[5], (const float*)d_in[9] };
  const float* b1[3] = { (const float*)d_in[2], (const float*)d_in[6], (const float*)d_in[10] };
  const float* w2[3] = { (const float*)d_in[3], (const float*)d_in[7], (const float*)d_in[11] };
  const float* b2[3] = { (const float*)d_in[4], (const float*)d_in[8], (const float*)d_in[12] };
  float* out = (float*)d_out;

  char* ws = (char*)d_ws;
  const size_t MB = (size_t)1 << 20;
  _Float16* xh  = (_Float16*)(ws + 0 * MB);    //  8 MiB: x in f16      [4096,1024]
  _Float16* wT1 = (_Float16*)(ws + 8 * MB);    //  8 MiB: W1^T f16      [4096,1024]
  _Float16* wT2 = (_Float16*)(ws + 16 * MB);   //  8 MiB: W2^T f16      [1024,4096]
  _Float16* hmid= (_Float16*)(ws + 24 * MB);   // 32 MiB: hidden f16    [4096,4096]
  _Float16* qh  = (_Float16*)(ws + 56 * MB);   //  8 MiB: Q f16         [4096,1024]
  _Float16* kh  = (_Float16*)(ws + 64 * MB);   //  8 MiB: K f16         [4096,1024]
  _Float16* vT  = (_Float16*)(ws + 72 * MB);   //  8 MiB: V^T f16       [B,D,S]

  const int M = BB * SS;       // 4096 tokens
  const int ntok_elems = M * DDIM;

  // x -> f16
  cvt_f32_f16_kernel<<<(ntok_elems + 255) / 256, 256, 0, stream>>>(x, xh, ntok_elems);

  _Float16* dst[3] = { qh, kh, vT };
  for (int p = 0; p < 3; ++p) {
    // W1 [1024,4096] -> W1^T [4096,1024] f16
    transpose_cvt_kernel<<<dim3(HIDD / 32, DDIM / 32), dim3(32, 8), 0, stream>>>(
        w1[p], wT1, DDIM, HIDD);
    // W2 [4096,1024] -> W2^T [1024,4096] f16
    transpose_cvt_kernel<<<dim3(DDIM / 32, HIDD / 32), dim3(32, 8), 0, stream>>>(
        w2[p], wT2, HIDD, DDIM);
    // h = silu(x @ W1 + b1)   (M=4096, N=4096, K=1024): 32x64 tiles -> 8192 waves
    gemm_silu_kernel<<<(M / 32) * (HIDD / 64) / 8, 256, 0, stream>>>(
        xh, wT1, b1[p], hmid, M, HIDD, DDIM);
    // y = h @ W2 + b2         (M=4096, N=1024, K=4096): 32x64 tiles -> 2048 waves
    gemm_out_kernel<<<(M / 32) * (DDIM / 64) / 8, 256, 0, stream>>>(
        hmid, wT2, b2[p], dst[p], M, DDIM, HIDD, (p == 2) ? 1 : 0);
  }

  // attention: one wave per 16 query rows: 2*16*128 = 4096 waves -> 512 blocks
  attn_kernel<<<(BB * NH * (SS / 16)) / 8, 256, 0, stream>>>(qh, kh, vT, out);
}
